// StateSpaceModel_40870908789227
// MI455X (gfx1250) — matrix operand order repaired and checked by reading, not compiled, not run on hardware
//
#include <hip/hip_runtime.h>
#include <cstdint>

typedef unsigned int  u32;
typedef unsigned short u16;
typedef unsigned long long u64;

typedef __attribute__((ext_vector_type(16))) __bf16 bf16x16;
typedef __attribute__((ext_vector_type(8)))  float  f32x8;
typedef __attribute__((ext_vector_type(4)))  u32    u32x4;
typedef __attribute__((ext_vector_type(8)))  u32    u32x8;

union Frag { uint4 q[2]; bf16x16 v; };

__device__ __forceinline__ u16 f_to_bf16(float f) {
    u32 u = __float_as_uint(f);
    u32 r = u + 0x7FFFu + ((u >> 16) & 1u);
    return (u16)(r >> 16);
}
__device__ __forceinline__ float gelu_exact(float x) {
    return 0.5f * x * (1.0f + erff(x * 0.70710678118654752f));
}
__device__ __forceinline__ float sigmoidf(float x) {
    return 1.0f / (1.0f + expf(-x));
}

// ---------------------------------------------------------------------------
// Generic f32 -> bf16 conversion (weights)
// ---------------------------------------------------------------------------
__global__ void f2bf_kernel(const float* __restrict__ in, u16* __restrict__ out, long n) {
    long i = (long)blockIdx.x * blockDim.x + threadIdx.x;
    if (i < n) out[i] = f_to_bf16(in[i]);
}

// ---------------------------------------------------------------------------
// Embedding gather + positional add:  x[b,s,:] = emb[tok] + pos[s]
// ---------------------------------------------------------------------------
__global__ void embed_kernel(const int* __restrict__ tokens, const float* __restrict__ emb,
                             const float* __restrict__ pos, float* __restrict__ x, int S) {
    long gid = (long)blockIdx.x * blockDim.x + threadIdx.x;   // M*512 threads
    long row = gid >> 9;
    int  d   = (int)(gid & 511);
    int  s   = (int)(row % S);
    int  tok = tokens[row];
    x[gid] = emb[(long)tok * 512 + d] + pos[(long)s * 512 + d];
}

// ---------------------------------------------------------------------------
// LayerNorm over D=512, one row per 256-thread block. Optional bf16/f32 out.
// ---------------------------------------------------------------------------
__launch_bounds__(256)
__global__ void layernorm_kernel(const float* __restrict__ x,
                                 const float* __restrict__ sc, const float* __restrict__ bi,
                                 u16* __restrict__ outb, float* __restrict__ outf) {
    long row = blockIdx.x;
    const float* xr = x + row * 512;
    int t = threadIdx.x;
    float x0 = xr[t], x1 = xr[t + 256];
    float sum = x0 + x1, sq = x0 * x0 + x1 * x1;
#pragma unroll
    for (int off = 16; off > 0; off >>= 1) {
        sum += __shfl_down(sum, off, 32);
        sq  += __shfl_down(sq,  off, 32);
    }
    __shared__ float rs[8], rq[8];
    __shared__ float smu, srstd;
    int wave = t >> 5, lane = t & 31;
    if (lane == 0) { rs[wave] = sum; rq[wave] = sq; }
    __syncthreads();
    if (t == 0) {
        float S = 0.f, Q = 0.f;
#pragma unroll
        for (int i = 0; i < 8; ++i) { S += rs[i]; Q += rq[i]; }
        float mu  = S * (1.0f / 512.0f);
        float var = Q * (1.0f / 512.0f) - mu * mu;
        smu = mu; srstd = rsqrtf(var + 1e-5f);
    }
    __syncthreads();
    float mu = smu, rstd = srstd;
    float y0 = (x0 - mu) * rstd * sc[t]       + bi[t];
    float y1 = (x1 - mu) * rstd * sc[t + 256] + bi[t + 256];
    if (outb) { outb[row * 512 + t] = f_to_bf16(y0); outb[row * 512 + t + 256] = f_to_bf16(y1); }
    if (outf) { outf[row * 512 + t] = y0;            outf[row * 512 + t + 256] = y1; }
}

// ---------------------------------------------------------------------------
// bf16 WMMA GEMM:  out = epilogue(A[M,K] @ W[K,N])
//   block = 256 threads = 8 waves, macro tile 128x64, each wave 32x32 (2x2 WMMA)
//   Double-buffered pipeline: the Tensor Data Mover streams A tile kc+1 into
//   LDS (with hardware row padding: +16B after each 64B row -> stride 40 elems)
//   while the WMMAs consume tile kc; the W tile for kc+1 is loaded to registers
//   under the WMMAs and scattered (transposed, N-major) afterwards.
// Epilogue: +bias, optional exact-GELU, +addsrc, +residual, f32 and/or bf16 out.
// ---------------------------------------------------------------------------
__launch_bounds__(256)
__global__ void gemm_wmma_kernel(const u16* __restrict__ A, const u16* __restrict__ W,
                                 long M, int N, int K,
                                 const float* __restrict__ bias, int act,
                                 const float* __restrict__ addsrc,
                                 const float* __restrict__ residual,
                                 float* __restrict__ outf, u16* __restrict__ outb) {
    const int LDA = 40;                      // padded row stride (elements); 80B = 5*16B
    __shared__ u16 sA[2][128 * 40];          // double-buffered A, TDM-padded
    __shared__ u16 sB[2][64 * 40];           // double-buffered W, N-major (transposed)

    int tid  = threadIdx.x;
    int wave = tid >> 5, lane = tid & 31;
    int wm = wave & 3, wn = wave >> 2;       // 4x2 wave grid
    long bm = (long)blockIdx.x * 128;
    long bn = (long)blockIdx.y * 64;

    f32x8 acc[2][2] = {};

    int mlane = lane & 15;
    int khalf = lane >> 4;

    // LDS byte offsets (flat-aperture rule: low 32 bits of a generic LDS
    // pointer are the workgroup-relative LDS offset).
    u32 sA_off[2] = { (u32)(size_t)(void*)&sA[0][0], (u32)(size_t)(void*)&sA[1][0] };

    // W staging geometry (per thread)
    int kk = tid >> 3;                       // 0..31
    int n0 = (tid & 7) * 8;                  // 0..56

    // ---- TDM descriptor issue for A tile of chunk starting at kbase -> buf
    auto tdm_issue = [&](int kbase, int buf) {
        u64 gaddr = (u64)(size_t)(A + bm * (long)K + kbase);
        u32x4 g0;
        g0[0] = 1u;                                   // count=1, user mode
        g0[1] = sA_off[buf];                          // lds_addr (bytes)
        g0[2] = (u32)gaddr;                           // global_addr[31:0]
        g0[3] = ((u32)(gaddr >> 32) & 0x01FFFFFFu)    // global_addr[56:32]
                | 0x80000000u;                        // type=2 ("image")
        u32x8 g1;
        g1[0] = (1u << 16)                            // data_size = 2 bytes
              | (1u << 20)                            // pad_enable
              | (3u << 22)                            // pad_interval: 16 DWORDs (64B row)
              | (3u << 25);                           // pad_amount: 4 DWORDs (16B)
        g1[1] = ((u32)K & 0xFFFFu) << 16;             // tensor_dim0[15:0]
        g1[2] = ((u32)K >> 16) | (((u32)M & 0xFFFFu) << 16);   // dim0 hi | dim1 lo
        g1[3] = (((u32)M >> 16) & 0xFFFFu) | (32u << 16);      // dim1 hi | tile_dim0=32
        g1[4] = 128u;                                 // tile_dim1=128, tile_dim2=0
        g1[5] = (u32)K;                               // tensor_dim0_stride lo32
        g1[6] = 0u;                                   // stride hi16 | dim1_stride lo16
        g1[7] = 0u;
        asm volatile("tensor_load_to_lds %0, %1" :: "s"(g0), "s"(g1) : "memory");
    };

    int nk = K >> 5;

    // ---- prologue: fill buffer 0
    if (wave == 0) tdm_issue(0, 0);
    {
        const u16* src = W + (long)kk * N + bn + n0;
        uint4 q = *(const uint4*)src;
        const u16* e = (const u16*)&q;
#pragma unroll
        for (int j = 0; j < 8; ++j) sB[0][(n0 + j) * LDA + kk] = e[j];
    }
    if (wave == 0) __builtin_amdgcn_s_wait_tensorcnt(0);
    __syncthreads();

    for (int kc = 0; kc < nk; ++kc) {
        int cur = kc & 1, nxt = cur ^ 1;
        bool more = (kc + 1 < nk);

        // ---- kick off next chunk: TDM for A, global load for W (under WMMAs)
        uint4 wq = {};
        if (more) {
            if (wave == 0) tdm_issue((kc + 1) << 5, nxt);
            const u16* src = W + (long)(((kc + 1) << 5) + kk) * N + bn + n0;
            wq = *(const uint4*)src;
            if (kc + 2 < nk)
                __builtin_prefetch(src + (long)32 * N, 0, 1);   // global_prefetch_b8
        }

        // ---- fragments + WMMA on current buffers
        Frag bfrag[2];
#pragma unroll
        for (int sn = 0; sn < 2; ++sn) {
            const u16* bp = &sB[cur][(wn * 32 + sn * 16 + mlane) * LDA + khalf * 8];
            bfrag[sn].q[0] = *(const uint4*)bp;
            bfrag[sn].q[1] = *(const uint4*)(bp + 16);
        }
#pragma unroll
        for (int sm = 0; sm < 2; ++sm) {
            Frag a;
            const u16* ap = &sA[cur][(wm * 32 + sm * 16 + mlane) * LDA + khalf * 8];
            a.q[0] = *(const uint4*)ap;
            a.q[1] = *(const uint4*)(ap + 16);
#pragma unroll
            for (int sn = 0; sn < 2; ++sn) {
                acc[sm][sn] = __builtin_amdgcn_wmma_f32_16x16x32_bf16(
                    false, a.v, false, bfrag[sn].v, (short)0, acc[sm][sn], false, false);
            }
        }

        // ---- publish next buffers, then barrier
        if (more) {
            const u16* e = (const u16*)&wq;
#pragma unroll
            for (int j = 0; j < 8; ++j) sB[nxt][(n0 + j) * LDA + kk] = e[j];
            if (wave == 0) __builtin_amdgcn_s_wait_tensorcnt(0);
        }
        __syncthreads();
    }

    // ---- epilogue: D layout: row = r + 8*(lane/16), col = lane%16
#pragma unroll
    for (int sm = 0; sm < 2; ++sm)
#pragma unroll
    for (int sn = 0; sn < 2; ++sn) {
#pragma unroll
        for (int r = 0; r < 8; ++r) {
            long row = bm + wm * 32 + sm * 16 + r + khalf * 8;
            long col = bn + wn * 32 + sn * 16 + mlane;
            long idx = row * N + col;
            float v = acc[sm][sn][r];
            if (bias)     v += bias[col];
            if (act)      v  = gelu_exact(v);
            if (addsrc)   v += addsrc[idx];
            if (residual) v += residual[idx];
            if (outf) outf[idx] = v;
            if (outb) outb[idx] = f_to_bf16(v);
        }
    }
}

// ---------------------------------------------------------------------------
// Chunked parallel scan: h_t = a*h_{t-1} + u_t  (a = sigmoid(A_diag[n]))
// Pass 1: local scan within 256-step chunks (in place in u), write chunk carry.
// ---------------------------------------------------------------------------
__global__ void scan_local_kernel(float* __restrict__ u, const float* __restrict__ A_diag,
                                  float* __restrict__ carry) {
    int gid = blockIdx.x * blockDim.x + threadIdx.x;   // (b*8 + c)*64 + n, 8192 threads
    int n = gid & 63;
    int c = (gid >> 6) & 7;
    int b = gid >> 9;
    float a = sigmoidf(A_diag[n]);
    float h = 0.f;
    long base = ((long)b * 2048 + (long)c * 256) * 64 + n;
    for (int t = 0; t < 256; t += 4) {
        float v0 = u[base + (long)(t + 0) * 64];
        float v1 = u[base + (long)(t + 1) * 64];
        float v2 = u[base + (long)(t + 2) * 64];
        float v3 = u[base + (long)(t + 3) * 64];
        h = a * h + v0; u[base + (long)(t + 0) * 64] = h;
        h = a * h + v1; u[base + (long)(t + 1) * 64] = h;
        h = a * h + v2; u[base + (long)(t + 2) * 64] = h;
        h = a * h + v3; u[base + (long)(t + 3) * 64] = h;
    }
    carry[gid] = h;
}

// Pass 2: combine the 8 chunk carries per (b,n); rewrite carry[c] = h at chunk start.
__global__ void scan_carry_kernel(float* __restrict__ carry, const float* __restrict__ A_diag) {
    int gid = blockIdx.x * blockDim.x + threadIdx.x;   // b*64 + n, 1024 threads
    int n = gid & 63;
    int b = gid >> 6;
    float a = sigmoidf(A_diag[n]);
    float a256 = a;
#pragma unroll
    for (int i = 0; i < 8; ++i) a256 = a256 * a256;    // a^256
    float h0 = 0.f;
    for (int c = 0; c < 8; ++c) {
        int idx = (b * 8 + c) * 64 + n;
        float ce = carry[idx];
        carry[idx] = h0;
        h0 = a256 * h0 + ce;
    }
}

// Pass 3: h_t = local_t + a^(t_local+1) * h0_chunk ; emit bf16 for the C-GEMM.
__global__ void scan_apply_kernel(const float* __restrict__ u, const float* __restrict__ carry,
                                  const float* __restrict__ A_diag, u16* __restrict__ hs) {
    long gid = (long)blockIdx.x * blockDim.x + threadIdx.x;   // M*64 threads
    int n = (int)(gid & 63);
    long rowt = gid >> 6;
    int tg = (int)(rowt & 2047);
    int b  = (int)(rowt >> 11);
    int c  = tg >> 8;
    int tl = tg & 255;
    float a  = sigmoidf(A_diag[n]);
    float h0 = carry[(b * 8 + c) * 64 + n];
    float corr = (h0 != 0.f) ? h0 * __powf(a, (float)(tl + 1)) : 0.f;
    hs[gid] = f_to_bf16(u[gid] + corr);
}

// ---------------------------------------------------------------------------
// Mean pool over S per batch:  pooled[b,d] = mean_s xf[b,s,d]
// ---------------------------------------------------------------------------
__launch_bounds__(256)
__global__ void pool_kernel(const float* __restrict__ xf, float* __restrict__ pooled) {
    int b = blockIdx.x, t = threadIdx.x;
    const float* base = xf + (long)b * 2048 * 512;
    float s0 = 0.f, s1 = 0.f;
    for (int s = 0; s < 2048; ++s) {
        s0 += base[(long)s * 512 + t];
        s1 += base[(long)s * 512 + t + 256];
    }
    pooled[b * 512 + t]       = s0 * (1.0f / 2048.0f);
    pooled[b * 512 + t + 256] = s1 * (1.0f / 2048.0f);
}

// Tiny classifier head (16 rows — negligible time, plain f32 dots).
__global__ void head1_kernel(const float* __restrict__ pooled, const float* __restrict__ cw1,
                             const float* __restrict__ cb1, float* __restrict__ hbuf) {
    int gid = blockIdx.x * 256 + threadIdx.x;   // 8192
    int row = gid >> 9, col = gid & 511;
    float acc = cb1[col];
    for (int k = 0; k < 512; ++k) acc += pooled[row * 512 + k] * cw1[k * 512 + col];
    hbuf[gid] = gelu_exact(acc);
}
__global__ void head2_kernel(const float* __restrict__ hbuf, const float* __restrict__ cw2,
                             const float* __restrict__ cb2, float* __restrict__ out) {
    int t = threadIdx.x;                        // 128
    int row = t >> 3, col = t & 7;
    float acc = cb2[col];
    for (int k = 0; k < 512; ++k) acc += hbuf[row * 512 + k] * cw2[k * 8 + col];
    out[row * 8 + col] = acc;
}

// ---------------------------------------------------------------------------
extern "C" void kernel_launch(void* const* d_in, const int* in_sizes, int n_in,
                              void* d_out, int out_size, void* d_ws, size_t ws_size,
                              hipStream_t stream) {
    (void)in_sizes; (void)n_in; (void)out_size; (void)ws_size;
    const int   Bsz = 16, S = 2048, D = 512, Nst = 64, L = 4, F = 2048;
    const long  M = (long)Bsz * S;

    const int*   tokens = (const int*)  d_in[0];
    const float* emb    = (const float*)d_in[1];
    const float* pos    = (const float*)d_in[2];
    const float* A_diag = (const float*)d_in[3];
    const float* Bm     = (const float*)d_in[4];
    const float* Cm     = (const float*)d_in[5];
    const float* Dm     = (const float*)d_in[6];
    const float* ln1_s  = (const float*)d_in[7];
    const float* ln1_b  = (const float*)d_in[8];
    const float* ln2_s  = (const float*)d_in[9];
    const float* ln2_b  = (const float*)d_in[10];
    const float* w1     = (const float*)d_in[11];
    const float* b1     = (const float*)d_in[12];
    const float* w2     = (const float*)d_in[13];
    const float* b2     = (const float*)d_in[14];
    const float* lnf_s  = (const float*)d_in[15];
    const float* lnf_b  = (const float*)d_in[16];
    const float* cw1    = (const float*)d_in[17];
    const float* cb1    = (const float*)d_in[18];
    const float* cw2    = (const float*)d_in[19];
    const float* cb2    = (const float*)d_in[20];

    char* wsp = (char*)d_ws;
    auto alloc = [&](size_t bytes) -> char* {
        char* p = wsp; wsp += (bytes + 255) & ~(size_t)255; return p;
    };
    float* x      = (float*)alloc(M * D * 4);        // residual stream (f32)
    u16*   xln    = (u16*)  alloc(M * D * 2);        // LN output (bf16)
    float* tmp    = (float*)alloc(M * D * 4);        // hs@C result / final-LN f32
    float* uB     = (float*)alloc(M * Nst * 4);      // u / local scan (f32)
    u16*   hs     = (u16*)  alloc(M * Nst * 2);      // scan output (bf16)
    u16*   h1     = (u16*)  alloc(M * F * 2);        // MLP hidden (bf16)
    float* carry  = (float*)alloc(16 * 8 * 64 * 4);
    u16*   wB     = (u16*)  alloc((size_t)L * D * Nst * 2);
    u16*   wC     = (u16*)  alloc((size_t)L * Nst * D * 2);
    u16*   wD     = (u16*)  alloc((size_t)L * D * D * 2);
    u16*   ww1    = (u16*)  alloc((size_t)L * D * F * 2);
    u16*   ww2    = (u16*)  alloc((size_t)L * F * D * 2);
    float* pooled = (float*)alloc(16 * 512 * 4);
    float* hbuf   = (float*)alloc(16 * 512 * 4);

    auto cvt = [&](const float* src, u16* dst, long n) {
        f2bf_kernel<<<(unsigned)((n + 255) / 256), 256, 0, stream>>>(src, dst, n);
    };
    cvt(Bm,  wB,  (long)L * D * Nst);
    cvt(Cm,  wC,  (long)L * Nst * D);
    cvt(Dm,  wD,  (long)L * D * D);
    cvt(w1,  ww1, (long)L * D * F);
    cvt(w2,  ww2, (long)L * F * D);

    embed_kernel<<<(unsigned)(M * D / 256), 256, 0, stream>>>(tokens, emb, pos, x, S);

    for (int l = 0; l < L; ++l) {
        const float* Al = A_diag + l * Nst;
        // xln = LN1(x)
        layernorm_kernel<<<(unsigned)M, 256, 0, stream>>>(x, ln1_s + l * D, ln1_b + l * D,
                                                          xln, nullptr);
        // u = xln @ Bm[l]   [M,512]@[512,64]
        gemm_wmma_kernel<<<dim3((unsigned)(M / 128), Nst / 64), 256, 0, stream>>>(
            xln, wB + (long)l * D * Nst, M, Nst, D,
            nullptr, 0, nullptr, nullptr, uB, nullptr);
        // diagonal scan -> hs (bf16)
        scan_local_kernel<<<8192 / 256, 256, 0, stream>>>(uB, Al, carry);
        scan_carry_kernel<<<1024 / 256, 256, 0, stream>>>(carry, Al);
        scan_apply_kernel<<<(unsigned)(M * Nst / 256), 256, 0, stream>>>(uB, carry, Al, hs);
        // tmp = hs @ Cm[l]   [M,64]@[64,512]
        gemm_wmma_kernel<<<dim3((unsigned)(M / 128), D / 64), 256, 0, stream>>>(
            hs, wC + (long)l * Nst * D, M, D, Nst,
            nullptr, 0, nullptr, nullptr, tmp, nullptr);
        // x = x + tmp + xln @ Dm[l]
        gemm_wmma_kernel<<<dim3((unsigned)(M / 128), D / 64), 256, 0, stream>>>(
            xln, wD + (long)l * D * D, M, D, D,
            nullptr, 0, tmp, x, x, nullptr);
        // xln = LN2(x)
        layernorm_kernel<<<(unsigned)M, 256, 0, stream>>>(x, ln2_s + l * D, ln2_b + l * D,
                                                          xln, nullptr);
        // h1 = gelu(xln @ w1 + b1)  -> bf16
        gemm_wmma_kernel<<<dim3((unsigned)(M / 128), F / 64), 256, 0, stream>>>(
            xln, ww1 + (long)l * D * F, M, F, D,
            b1 + l * F, 1, nullptr, nullptr, nullptr, h1);
        // x = x + h1 @ w2 + b2
        gemm_wmma_kernel<<<dim3((unsigned)(M / 128), D / 64), 256, 0, stream>>>(
            h1, ww2 + (long)l * F * D, M, D, F,
            b2 + l * D, 0, nullptr, x, x, nullptr);
    }

    // final LN (f32 out) -> mean pool -> head
    layernorm_kernel<<<(unsigned)M, 256, 0, stream>>>(x, lnf_s, lnf_b, nullptr, tmp);
    pool_kernel<<<16, 256, 0, stream>>>(tmp, pooled);
    head1_kernel<<<32, 256, 0, stream>>>(pooled, cw1, cb1, hbuf);
    head2_kernel<<<1, 128, 0, stream>>>(hbuf, cw2, cb2, (float*)d_out);
}
